// DIEN_77326591197307
// MI455X (gfx1250) — matrix operand order repaired and checked by reading, not compiled, not run on hardware
//
#include <hip/hip_runtime.h>

// CDNA5 / gfx1250, wave32. Fused AUGRU scan:
//   one wave per 16 batch rows, h kept in registers across T=200 steps,
//   all matmuls via v_wmma_f32_16x16x32_f16.
// Round-3 reasoning: the scan is latency/issue bound at <=1 wave per SIMD.
//   * Weights register-resident (48 B-tiles = 384 VGPRs, gfx1250 has 1024 +
//     MSB addressing): s_set_vgpr_msb costs issue slots but never stalls,
//     whereas per-step LDS weight loads produced a ds_load->s_wait_dscnt 0
//     ->wmma ladder (a full LDS latency per WMMA, round-2 asm).
//   * Lean elementwise tail from round 2: accumulators start at inline-0,
//     bias folded into hard-sigmoid constants, mask folded into the gate,
//     v_med3_f32 + v_tanh_f32.
//   * Only per-step LDS traffic is the h C-layout -> A-layout transpose.

typedef __attribute__((ext_vector_type(16))) _Float16 v16h;
typedef __attribute__((ext_vector_type(8)))  float    v8f;

#define T_LEN 200
#define D_DIM 64
#define U_DIM 64
#define G3    192   // 3*U
#define NT3   12    // G3/16

__device__ __forceinline__ v8f wmma16(v16h a, v16h b, v8f c) {
  // (neg_a, A, neg_b, B, c_mod, C, reuse_a, reuse_b)
  return __builtin_amdgcn_wmma_f32_16x16x32_f16(false, a, false, b, (short)0, c,
                                                false, false);
}

// hard_sigmoid with pre-folded bias: clip(0.2*(x+b)+0.5, 0, 1)
//   = med3(fma(x, 0.2, c), 0, 1) with c = 0.2*b + 0.5
__device__ __forceinline__ float hsig_fold(float x, float c) {
  return __builtin_amdgcn_fmed3f(fmaf(x, 0.2f, c), 0.0f, 1.0f);
}

__device__ __forceinline__ float fast_tanh(float x) {
#if __has_builtin(__builtin_amdgcn_tanhf)
  return __builtin_amdgcn_tanhf(x);
#elif __has_builtin(__builtin_amdgcn_tanh_f32)
  return __builtin_amdgcn_tanh_f32(x);
#else
  return tanhf(x);
#endif
}

__device__ __forceinline__ v16h cvt16(float4 f0, float4 f1, float4 f2, float4 f3) {
  v16h a;
  a[0]  = (_Float16)f0.x; a[1]  = (_Float16)f0.y; a[2]  = (_Float16)f0.z; a[3]  = (_Float16)f0.w;
  a[4]  = (_Float16)f1.x; a[5]  = (_Float16)f1.y; a[6]  = (_Float16)f1.z; a[7]  = (_Float16)f1.w;
  a[8]  = (_Float16)f2.x; a[9]  = (_Float16)f2.y; a[10] = (_Float16)f2.z; a[11] = (_Float16)f2.w;
  a[12] = (_Float16)f3.x; a[13] = (_Float16)f3.y; a[14] = (_Float16)f3.z; a[15] = (_Float16)f3.w;
  return a;
}

// A-operand (16x32 f16) loader per ISA 7.12.2: lane L holds row L%16; lanes
// 0-15 take K = {cbase..+7, cbase+16..+23}, lanes 16-31 the +8 offsets.
__device__ __forceinline__ v16h loadA(const float* row, int cbase, int grp) {
  const float4* p0 = (const float4*)(row + cbase + grp * 8);
  const float4* p1 = (const float4*)(row + cbase + 16 + grp * 8);
  return cvt16(p0[0], p0[1], p1[0], p1[1]);
}

__global__ __launch_bounds__(32) void augru_fused(
    const float* __restrict__ inputs,       // [B, T, D]
    const float* __restrict__ alphas,       // [B, T]
    const unsigned char* __restrict__ mask, // [B, T] bool (1 byte)
    const float* __restrict__ kernelW,      // [D, 3U]
    const float* __restrict__ rkernelW,     // [U, 3U]
    const float* __restrict__ bias,         // [2, 3U]
    float* __restrict__ out)                // [B, U]
{
  // B-operand-layout weight staging (f16) + h transpose buffer.
  __shared__ alignas(32) _Float16 wlds[2][2][NT3][32][16]; // 48 KB
  __shared__ alignas(16) float    hbuf[16 * U_DIM];        // 4 KB

  const int lane = threadIdx.x & 31;
  const int grp  = lane >> 4;   // 0: lanes 0-15, 1: lanes 16-31
  const int lc   = lane & 15;   // tile column (B/C layout) or tile row (A layout)
  const int b0   = blockIdx.x * 16;

  // ---- stage weights into WMMA B-operand layout ----
  // B tile (K-chunk c, N-tile nt): lane holds column N = nt*16+lc, with K =
  // c*32 + {grp*8..+7, 16+grp*8..+7} (matches A-operand table order).
  for (int w = 0; w < 2; ++w) {
    const float* W = w ? rkernelW : kernelW;
    for (int c = 0; c < 2; ++c) {
      for (int nt = 0; nt < NT3; ++nt) {
        const int kbase = c * 32 + grp * 8;
        const int col   = nt * 16 + lc;
        v16h v;
#pragma unroll
        for (int e = 0; e < 16; ++e)
          v[e] = (_Float16)W[(kbase + (e < 8 ? e : e + 8)) * G3 + col];
        *(v16h*)&wlds[w][c][nt][lane][0] = v;
      }
    }
  }

  // per-lane gate constants (column = j*16 + lc).
  // z/r: bias (input+recurrent) folded into the hard-sigmoid fma constant.
  // candidate: keep raw input/recurrent bias for xh / rh.
  float czs[4], crs[4], bxh[4], brh[4];
#pragma unroll
  for (int j = 0; j < 4; ++j) {
    const int cz = j * 16 + lc, cr = 64 + j * 16 + lc, ch = 128 + j * 16 + lc;
    czs[j] = fmaf(bias[cz] + bias[G3 + cz], 0.2f, 0.5f);
    crs[j] = fmaf(bias[cr] + bias[G3 + cr], 0.2f, 0.5f);
    bxh[j] = bias[ch];
    brh[j] = bias[G3 + ch];
  }

  __syncthreads();

  // ---- preload all 48 weight B-operands into registers (~384 VGPRs) ----
  v16h wreg[2][2][NT3];
#pragma unroll
  for (int w = 0; w < 2; ++w)
#pragma unroll
    for (int c = 0; c < 2; ++c)
#pragma unroll
      for (int nt = 0; nt < NT3; ++nt)
        wreg[w][c][nt] = *(const v16h*)&wlds[w][c][nt][lane][0];

  // h state in WMMA C layout: h[j][i] = h[row i+8*grp][col j*16+lc]
  v8f h[4];
#pragma unroll
  for (int j = 0; j < 4; ++j)
#pragma unroll
    for (int i = 0; i < 8; ++i) h[j][i] = 0.0f;

  v16h hA0, hA1;  // h as A operand (h0 = 0)
#pragma unroll
  for (int e = 0; e < 16; ++e) { hA0[e] = (_Float16)0.0f; hA1[e] = (_Float16)0.0f; }

  const float* xbase = inputs + (size_t)(b0 + lc) * T_LEN * D_DIM;
  const int rb = b0 + grp * 8;

#pragma unroll 1
  for (int t = 0; t < T_LEN; ++t) {
    // x tile as two A operands (rows b0..b0+15, K = D)
    const float* xr = xbase + t * D_DIM;
    const v16h xA0 = loadA(xr, 0, grp);
    const v16h xA1 = loadA(xr, 32, grp);

    // per-row attention, with mask pre-folded: amv = m ? alpha : 0
    float amv[8];
#pragma unroll
    for (int i = 0; i < 8; ++i) {
      const float a = alphas[(size_t)(rb + i) * T_LEN + t];
      amv[i] = mask[(size_t)(rb + i) * T_LEN + t] ? a : 0.0f;
    }

#pragma unroll
    for (int j = 0; j < 4; ++j) {
      // accumulators start at inline 0 (bias folded into gate constants)
      v8f accZ = {}, accR = {}, accXH = {}, accRH = {};

      // z gate: xz + rz
      accZ = wmma16(hA0, wreg[1][0][j], accZ);
      accZ = wmma16(hA1, wreg[1][1][j], accZ);
      accZ = wmma16(xA0, wreg[0][0][j], accZ);
      accZ = wmma16(xA1, wreg[0][1][j], accZ);
      // r gate: xr + rr
      accR = wmma16(hA0, wreg[1][0][4 + j], accR);
      accR = wmma16(hA1, wreg[1][1][4 + j], accR);
      accR = wmma16(xA0, wreg[0][0][4 + j], accR);
      accR = wmma16(xA1, wreg[0][1][4 + j], accR);
      // candidate: xh and rh kept separate (hh = tanh(xh + r*rh))
      accXH = wmma16(xA0, wreg[0][0][8 + j], accXH);
      accXH = wmma16(xA1, wreg[0][1][8 + j], accXH);
      accRH = wmma16(hA0, wreg[1][0][8 + j], accRH);
      accRH = wmma16(hA1, wreg[1][1][8 + j], accRH);

#pragma unroll
      for (int i = 0; i < 8; ++i) {
        const float z  = amv[i] * hsig_fold(accZ[i], czs[j]);     // mask folded in
        const float r  = hsig_fold(accR[i], crs[j]);
        const float hh = fast_tanh(fmaf(r, accRH[i] + brh[j], accXH[i] + bxh[j]));
        h[j][i] = fmaf(z, hh - h[j][i], h[j][i]);  // h + m*z*(hh - h)
      }
    }

    // C-layout -> A-layout transpose of h through LDS (DS ops are in-order
    // within a wave; single-wave block, no barrier needed).
#pragma unroll
    for (int j = 0; j < 4; ++j)
#pragma unroll
      for (int i = 0; i < 8; ++i)
        hbuf[(i + grp * 8) * U_DIM + j * 16 + lc] = h[j][i];

    const float* hr = &hbuf[lc * U_DIM];
    hA0 = loadA(hr, 0, grp);
    hA1 = loadA(hr, 32, grp);
  }

  // write last_h [B, U]
#pragma unroll
  for (int j = 0; j < 4; ++j)
#pragma unroll
    for (int i = 0; i < 8; ++i)
      out[(size_t)(rb + i) * U_DIM + j * 16 + lc] = h[j][i];
}

extern "C" void kernel_launch(void* const* d_in, const int* in_sizes, int n_in,
                              void* d_out, int out_size, void* d_ws, size_t ws_size,
                              hipStream_t stream) {
  const float* inputs   = (const float*)d_in[0];
  const float* alphas   = (const float*)d_in[1];
  const unsigned char* mask = (const unsigned char*)d_in[2]; // jnp bool = 1 byte
  const float* kernelW  = (const float*)d_in[3];
  const float* rkernelW = (const float*)d_in[4];
  const float* bias     = (const float*)d_in[5];
  float* out = (float*)d_out;

  const int B = in_sizes[1] / T_LEN;  // alphas is [B, T]
  dim3 grid(B / 16), block(32);
  hipLaunchKernelGGL(augru_fused, grid, block, 0, stream,
                     inputs, alphas, mask, kernelW, rkernelW, bias, out);
}
